// AbstractLinear_85718957293609
// MI455X (gfx1250) — matrix operand order repaired and checked by reading, not compiled, not run on hardware
//
#include <hip/hip_runtime.h>
#include <hip/hip_bf16.h>

#define BATCH 4096
#define IN    4096
#define OUT   16384

typedef unsigned short u16;
typedef __attribute__((ext_vector_type(16))) __bf16 v16bf;
typedef __attribute__((ext_vector_type(8)))  float  v8f;
typedef __attribute__((ext_vector_type(4)))  unsigned int v4u;

union Frag { v16bf f; v4u q[2]; };

// ---------------------------------------------------------------------------
// Split fp32 -> bf16 hi/lo pair (x = hi + lo to ~16 mantissa bits).
// Vectorized: 4 elements / thread.
// ---------------------------------------------------------------------------
__global__ __launch_bounds__(256) void split_bf16_kernel(
    const float* __restrict__ src, u16* __restrict__ hi, u16* __restrict__ lo,
    int n4)
{
    int i = blockIdx.x * blockDim.x + threadIdx.x;
    if (i >= n4) return;
    float4 v = ((const float4*)src)[i];
    float f[4] = {v.x, v.y, v.z, v.w};
    unsigned int h[4], l[4];
#pragma unroll
    for (int j = 0; j < 4; ++j) {
        __bf16 bh = (__bf16)f[j];                 // RTNE to bf16
        float  rem = f[j] - (float)bh;
        __bf16 bl = (__bf16)rem;
        h[j] = (unsigned int)__builtin_bit_cast(unsigned short, bh);
        l[j] = (unsigned int)__builtin_bit_cast(unsigned short, bl);
    }
    uint2 ho, loo;
    ho.x  = h[0] | (h[1] << 16);  ho.y  = h[2] | (h[3] << 16);
    loo.x = l[0] | (l[1] << 16);  loo.y = l[2] | (l[3] << 16);
    ((uint2*)hi)[i] = ho;
    ((uint2*)lo)[i] = loo;
}

// ---------------------------------------------------------------------------
// GEMM: y[b][o] = sum_k x[b][k] * W[o][k] + bias[o], bf16x3 (hi/lo split).
// Block: 256 threads = 8 waves in a 2(M) x 4(N) wave grid.
// Block tile 64(M) x 256(N); wave tile 32 x 64 = 2x4 WMMA 16x16 tiles; K=32.
// Grid: x = batch tiles, y = out tiles  ->  consecutive blocks share one
// 4.2 MB W panel (HBM once, then L2) while the 67 MB bf16 x stays
// L2-resident (192 MB L2).
// LDS tiles are K-contiguous with row stride 40 bf16 (80B): 16B-aligned,
// bank-conflict-free b128 fragment loads (20 banks/row step, gcd(20,64)=4,
// 16 distinct bank groups over 16 lanes).
// ---------------------------------------------------------------------------
__global__ __launch_bounds__(256) void gemm_bf16x3_kernel(
    const u16* __restrict__ xhi, const u16* __restrict__ xlo,
    const u16* __restrict__ whi, const u16* __restrict__ wlo,
    const float* __restrict__ bias, float* __restrict__ y)
{
    __shared__ __align__(16) u16 sXhi[64 * 40];
    __shared__ __align__(16) u16 sXlo[64 * 40];
    __shared__ __align__(16) u16 sWhi[256 * 40];
    __shared__ __align__(16) u16 sWlo[256 * 40];

    const int bm0  = blockIdx.x * 64;    // batch tile
    const int bn0  = blockIdx.y * 256;   // out tile
    const int t    = threadIdx.x;
    const int lane = t & 31;
    const int wave = t >> 5;
    const int wm   = wave & 1;   // 0..1  -> M offset wm*32
    const int wn   = wave >> 1;  // 0..3  -> N offset wn*64

    v8f acc[2][4] = {};

    // global->LDS mapping: x tile 64x32 (1 x b128/thread/array),
    //                      W tile 256x32 (4 x b128/thread/array)
    const int xrow = t >> 2;            // 0..63
    const int xko  = (t & 3) * 8;       // 0,8,16,24

    // fragment addressing (ISA 7.12.2 layouts, wave32)
    const int half  = lane >> 4;        // lanes 0-15 vs 16-31
    const int row15 = lane & 15;
    const int aoff0 = half * 8;         // A: K 0..7 / 8..15 (then +16)
    const int boff0 = half * 8;         // B: K 0..7 / 8..15 (then +16)

    for (int k0 = 0; k0 < IN; k0 += 32) {
        __syncthreads();
        {
            size_t gx = (size_t)(bm0 + xrow) * IN + k0 + xko;
            *(v4u*)&sXhi[xrow * 40 + xko] = *(const v4u*)(xhi + gx);
            *(v4u*)&sXlo[xrow * 40 + xko] = *(const v4u*)(xlo + gx);
            __builtin_prefetch(xhi + gx + 32, 0, 1);   // global_prefetch_b8
            __builtin_prefetch(xlo + gx + 32, 0, 1);
#pragma unroll
            for (int i = 0; i < 4; ++i) {
                int c  = t + i * 256;
                int r  = c >> 2;            // 0..255
                int ko = (c & 3) * 8;
                size_t gw = (size_t)(bn0 + r) * IN + k0 + ko;
                *(v4u*)&sWhi[r * 40 + ko] = *(const v4u*)(whi + gw);
                *(v4u*)&sWlo[r * 40 + ko] = *(const v4u*)(wlo + gw);
                if (i == 0) {
                    __builtin_prefetch(whi + gw + 32, 0, 1);
                    __builtin_prefetch(wlo + gw + 32, 0, 1);
                }
            }
        }
        __syncthreads();

        Frag ah[2], al[2], bh[4], bl[4];
#pragma unroll
        for (int mi = 0; mi < 2; ++mi) {
            int r = (wm * 32 + mi * 16 + row15) * 40;
            ah[mi].q[0] = *(const v4u*)&sXhi[r + aoff0];
            ah[mi].q[1] = *(const v4u*)&sXhi[r + aoff0 + 16];
            al[mi].q[0] = *(const v4u*)&sXlo[r + aoff0];
            al[mi].q[1] = *(const v4u*)&sXlo[r + aoff0 + 16];
        }
#pragma unroll
        for (int ni = 0; ni < 4; ++ni) {
            int r = (wn * 64 + ni * 16 + row15) * 40;
            bh[ni].q[0] = *(const v4u*)&sWhi[r + boff0];
            bh[ni].q[1] = *(const v4u*)&sWhi[r + boff0 + 16];
            bl[ni].q[0] = *(const v4u*)&sWlo[r + boff0];
            bl[ni].q[1] = *(const v4u*)&sWlo[r + boff0 + 16];
        }
#pragma unroll
        for (int mi = 0; mi < 2; ++mi) {
#pragma unroll
            for (int ni = 0; ni < 4; ++ni) {
                acc[mi][ni] = __builtin_amdgcn_wmma_f32_16x16x32_bf16(
                    false, ah[mi].f, false, bh[ni].f, (short)0, acc[mi][ni], false, false);
                acc[mi][ni] = __builtin_amdgcn_wmma_f32_16x16x32_bf16(
                    false, ah[mi].f, false, bl[ni].f, (short)0, acc[mi][ni], false, false);
                acc[mi][ni] = __builtin_amdgcn_wmma_f32_16x16x32_bf16(
                    false, al[mi].f, false, bh[ni].f, (short)0, acc[mi][ni], false, false);
            }
        }
    }

    // epilogue: C/D layout -> VGPR r holds (M = r + 8*half, N = lane&15)
#pragma unroll
    for (int ni = 0; ni < 4; ++ni) {
        int col = bn0 + wn * 64 + ni * 16 + row15;
        float bv = bias[col];
#pragma unroll
        for (int mi = 0; mi < 2; ++mi) {
#pragma unroll
            for (int r = 0; r < 8; ++r) {
                int row = bm0 + wm * 32 + mi * 16 + half * 8 + r;
                y[(size_t)row * OUT + col] = acc[mi][ni][r] + bv;
            }
        }
    }
}

// ---------------------------------------------------------------------------
// Interval bound propagation: one wave (32 lanes) per output row, fp32.
//   mid = W[o,:]·center + b[o],  rad = |W[o,:]|·radius
//   low_out = mid - rad, high_out = mid + rad
// center/radius derived on the fly; low/high stay L2-resident (32 KB).
// ---------------------------------------------------------------------------
__global__ __launch_bounds__(256) void ibp_kernel(
    const float* __restrict__ W, const float* __restrict__ low,
    const float* __restrict__ high, const float* __restrict__ bias,
    float* __restrict__ low_out, float* __restrict__ high_out)
{
    const int wave = threadIdx.x >> 5;
    const int lane = threadIdx.x & 31;
    const int o = blockIdx.x * 8 + wave;

    const float4* Wr  = (const float4*)(W + (size_t)o * IN);
    const float4* lo4 = (const float4*)low;
    const float4* hi4 = (const float4*)high;

    float accm = 0.f, accr = 0.f;
#pragma unroll 4
    for (int it = 0; it < IN / 128; ++it) {
        int k4 = it * 32 + lane;
        float4 w = Wr[k4];
        float4 l = lo4[k4];
        float4 h = hi4[k4];
        float c0 = 0.5f * (l.x + h.x), r0 = 0.5f * (h.x - l.x);
        float c1 = 0.5f * (l.y + h.y), r1 = 0.5f * (h.y - l.y);
        float c2 = 0.5f * (l.z + h.z), r2 = 0.5f * (h.z - l.z);
        float c3 = 0.5f * (l.w + h.w), r3 = 0.5f * (h.w - l.w);
        accm += w.x * c0 + w.y * c1 + w.z * c2 + w.w * c3;
        accr += fabsf(w.x) * r0 + fabsf(w.y) * r1 + fabsf(w.z) * r2 + fabsf(w.w) * r3;
    }
#pragma unroll
    for (int off = 16; off > 0; off >>= 1) {
        accm += __shfl_down(accm, off, 32);
        accr += __shfl_down(accr, off, 32);
    }
    if (lane == 0) {
        float mid = accm + bias[o];
        low_out[o]  = mid - accr;
        high_out[o] = mid + accr;
    }
}

// ---------------------------------------------------------------------------
extern "C" void kernel_launch(void* const* d_in, const int* in_sizes, int n_in,
                              void* d_out, int out_size, void* d_ws, size_t ws_size,
                              hipStream_t stream)
{
    (void)in_sizes; (void)n_in; (void)out_size; (void)ws_size;

    const float* x    = (const float*)d_in[0];
    const float* low  = (const float*)d_in[1];
    const float* high = (const float*)d_in[2];
    const float* W    = (const float*)d_in[3];
    const float* b    = (const float*)d_in[4];

    float* y        = (float*)d_out;
    float* low_out  = y + (size_t)BATCH * OUT;
    float* high_out = low_out + OUT;

    const size_t XN = (size_t)BATCH * IN;   // 16.8M
    const size_t WN = (size_t)OUT * IN;     // 67.1M
    u16* xhi = (u16*)d_ws;
    u16* xlo = xhi + XN;
    u16* whi = xlo + XN;
    u16* wlo = whi + WN;

    const int xn4 = (int)(XN / 4);
    split_bf16_kernel<<<(xn4 + 255) / 256, 256, 0, stream>>>(x, xhi, xlo, xn4);
    const int wn4 = (int)(WN / 4);
    split_bf16_kernel<<<(wn4 + 255) / 256, 256, 0, stream>>>(W, whi, wlo, wn4);

    dim3 grid(BATCH / 64, OUT / 256);   // x = batch tiles (L2: share W panel)
    gemm_bf16x3_kernel<<<grid, 256, 0, stream>>>(xhi, xlo, whi, wlo, b, y);

    ibp_kernel<<<OUT / 8, 256, 0, stream>>>(W, low, high, b, low_out, high_out);
}
